// GAT_20907900797471
// MI455X (gfx1250) — compile-verified, matrix operand order checked by reading, not run on hardware
//
#include <hip/hip_runtime.h>

// ---------------------------------------------------------------------------
// GAT (3 layers) for MI455X / gfx1250, wave32 + WMMA + TDM.
//   N=100000 nodes, DEG=16 (dst contiguous: edges of node n = [n*16, n*16+16)),
//   IN=128, H=4 heads, D=16 hidden, C=47 classes.
// fs = h@Wsrc via v_wmma_f32_16x16x32_f16 (f16 in, f32 acc); packed B staged
// into LDS once per block via tensor_load_to_lds (TDM, 6-arg clang-23 form).
// el/er computed with collapsed weights (Wsrc·a_l, Wdst·a_r) -> no fd GEMM.
// ---------------------------------------------------------------------------

#define NN      100000
#define DEG     16
#define NH      4
#define NEG_SLOPE 0.2f

typedef float    v8f  __attribute__((ext_vector_type(8)));
typedef _Float16 v16h __attribute__((ext_vector_type(16)));
typedef _Float16 v8h  __attribute__((ext_vector_type(8)));
typedef unsigned int u32x4 __attribute__((ext_vector_type(4)));
typedef int          i32x8 __attribute__((ext_vector_type(8)));
typedef int          i32x4 __attribute__((ext_vector_type(4)));

#if __has_builtin(__builtin_amdgcn_tensor_load_to_lds)
#define HAVE_TDM 1
#else
#define HAVE_TDM 0
#endif

// ---------------------------- f32 -> f16 convert ---------------------------
__global__ void gat_f32_to_f16(const float* __restrict__ in,
                               _Float16* __restrict__ out, int count) {
  int i = blockIdx.x * blockDim.x + threadIdx.x;
  if (i < count) out[i] = (_Float16)in[i];
}

// ------------------- pack W (f32 [fin x cols]) into WMMA-B layout ----------
// B (32x16 f16 per k-step): lane L holds column n = nt*16 + (L&15);
// its 16 halves hold K = kstep*32 + 8*(L>>4) + (j&7) + 16*(j>>3).
// Bp flat index == thread index (nt, kstep, lane, j). Pads cols with zeros.
__global__ void gat_pack_b(const float* __restrict__ W, _Float16* __restrict__ Bp,
                           int ksteps, int ntiles, int cols) {
  int t = blockIdx.x * blockDim.x + threadIdx.x;
  int total = ntiles * ksteps * 512;
  if (t >= total) return;
  int j    = t & 15;
  int lane = (t >> 4) & 31;
  int kt   = (t >> 9) % ksteps;
  int nt   = (t >> 9) / ksteps;
  int n = nt * 16 + (lane & 15);
  int K = kt * 32 + ((lane >> 4) << 3) + (j & 7) + ((j >> 3) << 4);
  float v = (n < cols) ? W[(size_t)K * cols + n] : 0.f;
  Bp[t] = (_Float16)v;
}

// -------- collapsed attention weights: wv[k][h] = sum_d W[k, h*dout+d]*a[h,d]
__global__ void gat_make_w(const float* __restrict__ W, const float* __restrict__ avec,
                           float* __restrict__ wv, int fin, int dout) {
  int t = blockIdx.x * blockDim.x + threadIdx.x;
  if (t >= fin * NH) return;
  int k = t >> 2, h = t & 3;
  int cols = NH * dout;
  float s = 0.f;
  for (int d = 0; d < dout; ++d)
    s += W[(size_t)k * cols + h * dout + d] * avec[h * dout + d];
  wv[k * NH + h] = s;
}

// ----------------- el/er: out[n,h] = h_in[n,:] . wv[:,h] -------------------
__global__ void gat_edge_logit(const float* __restrict__ hin, const float* __restrict__ wv,
                               float* __restrict__ out, int n_nodes, int fin) {
  int t = blockIdx.x * blockDim.x + threadIdx.x;
  if (t >= n_nodes * NH) return;
  int n = t >> 2, h = t & 3;
  const float* r = hin + (size_t)n * fin;
  float s = 0.f;
  for (int k = 0; k < fin; ++k) s += r[k] * wv[k * NH + h];
  out[t] = s;
}

// ----------------- TDM: linear copy of n8*8 bytes global -> LDS ------------
// D# per CDNA5 ISA §8.3/8.4: group0 = {count=1, lds_addr, global_addr, type=2},
// group1 = {mask=0, data_size=8B, tensor_dim0=n8, tensor_dim1=1, tile_dim0=n8,
//           tensor_dim0_stride=n8}; groups 2/3 zero (<=2D tensor).
// clang-23 6-arg builtin: (u32x4, i32x8, i32x4, i32x4, i32x8, i32 cpol).
#if HAVE_TDM
__device__ __forceinline__ void tdm_copy_to_lds(void* lds_dst, const void* gsrc,
                                                unsigned n8) {
  unsigned lds_off = (unsigned)(unsigned long long)(uintptr_t)lds_dst; // low 32b = LDS offset
  unsigned long long ga = (unsigned long long)(uintptr_t)gsrc;
  u32x4 g0;
  g0[0] = 1u;                                             // count=1
  g0[1] = lds_off;                                        // lds_addr
  g0[2] = (unsigned)(ga & 0xFFFFFFFFu);                   // global_addr[31:0]
  g0[3] = (unsigned)((ga >> 32) & 0x01FFFFFFu) | (2u << 30); // addr[56:32], type=2
  i32x8 g1;
  g1[0] = 0x00030000;                         // workgroup_mask=0, data_size=3 (8B)
  g1[1] = (int)((n8 & 0xFFFFu) << 16);        // tensor_dim0[15:0]
  g1[2] = (int)((n8 >> 16) | (1u << 16));     // tensor_dim0[31:16], tensor_dim1=1
  g1[3] = (int)((n8 & 0xFFFFu) << 16);        // tile_dim0 = n8
  g1[4] = 0;                                  // tile_dim1=0, tile_dim2=0
  g1[5] = (int)n8;                            // tensor_dim0_stride[31:0]
  g1[6] = 0;
  g1[7] = 0;
  i32x4 z4 = {0, 0, 0, 0};
  i32x8 z8 = {0, 0, 0, 0, 0, 0, 0, 0};
  __builtin_amdgcn_tensor_load_to_lds(g0, g1, z4, z4, z8, 0);
}
#endif

// ------------------------------- WMMA GEMM ---------------------------------
// C[n_rows x colsP] = A16[n_rows x (KSTEPS*32)] * Bp  (one wave per 16-row tile)
// A 16x32 f16 layout: lane L holds row M = L&15; 16 halves hold
//   K = 8*(L>>4) + (j&7) + 16*(j>>3)   (two contiguous 16B chunks per k-step).
// D 16x16 f32 layout: lane L holds col n = L&15; VGPR r holds row r + 8*(L>>4).
// Packed B (<=24KB) staged into LDS once per block (TDM when available).
template <int KSTEPS>
__global__ __launch_bounds__(256)
void gat_gemm_wmma(const _Float16* __restrict__ A, const _Float16* __restrict__ Bp,
                   float* __restrict__ C, int mtiles, int ntiles, int colsP) {
  __shared__ _Float16 sB[12288];                  // 24KB: max ntiles*ksteps*512
  const unsigned n8 = (unsigned)(ntiles * KSTEPS) * 128u; // 8-byte units
#if HAVE_TDM
  if (threadIdx.x < 32) {                         // one wave issues the DMA
    tdm_copy_to_lds(sB, Bp, n8);
    __builtin_amdgcn_s_wait_tensorcnt(0);
  }
#else
  for (unsigned i = threadIdx.x; i < n8; i += blockDim.x)
    ((unsigned long long*)sB)[i] = ((const unsigned long long*)Bp)[i];
#endif
  __syncthreads();

  const int fin  = KSTEPS * 32;
  int wave = (blockIdx.x * blockDim.x + threadIdx.x) >> 5;
  int lane = threadIdx.x & 31;
  if (wave >= mtiles) return;          // wave-uniform: EXEC stays all-ones
  int row = wave * 16 + (lane & 15);   // mtiles*16 == n_rows (N % 16 == 0)
  const int base = (lane >> 4) * 8;
  const _Float16* arow = A + (size_t)row * fin;

  v16h a[KSTEPS];
#pragma unroll
  for (int k = 0; k < KSTEPS; ++k) {
    v8h lo = *(const v8h*)(arow + k * 32 + base);
    v8h hi = *(const v8h*)(arow + k * 32 + base + 16);
    a[k] = __builtin_shufflevector(lo, hi, 0, 1, 2, 3, 4, 5, 6, 7,
                                   8, 9, 10, 11, 12, 13, 14, 15);
  }

  for (int nt = 0; nt < ntiles; ++nt) {
    v8f c = {};
#pragma unroll
    for (int k = 0; k < KSTEPS; ++k) {
      const _Float16* bp = sB + (((size_t)nt * KSTEPS + k) * 32 + lane) * 16;
      v8h blo = *(const v8h*)bp;
      v8h bhi = *(const v8h*)(bp + 8);
      v16h b = __builtin_shufflevector(blo, bhi, 0, 1, 2, 3, 4, 5, 6, 7,
                                       8, 9, 10, 11, 12, 13, 14, 15);
      c = __builtin_amdgcn_wmma_f32_16x16x32_f16(false, a[k], false, b,
                                                 (short)0, c, false, false);
    }
    float* crow = C + (size_t)(wave * 16 + 8 * (lane >> 4)) * colsP
                    + nt * 16 + (lane & 15);
#pragma unroll
    for (int r = 0; r < 8; ++r) crow[(size_t)r * colsP] = c[r];
  }
}

// --------------------- edge softmax + aggregation --------------------------
// Block handles `nb` nodes. Phase 1: nb*NH threads do the 16-edge
// leaky-relu + softmax into LDS (+ prefetch of gathered fs rows).
// Phase 2: nb*cols_real threads gather alpha * fs[src] (coalesced over
// channels), add bias, optional relu.
__global__ __launch_bounds__(256)
void gat_attn_agg(const int* __restrict__ src, const float* __restrict__ el,
                  const float* __restrict__ er, const float* __restrict__ fs,
                  const float* __restrict__ bias, float* __restrict__ out,
                  int n_nodes, int nb, int cols_real, int dout, int colsP,
                  int do_relu) {
  __shared__ float s_alpha[4 * NH * DEG];
  __shared__ int   s_src[4 * DEG];
  int node0 = blockIdx.x * nb;
  int t = threadIdx.x;

  if (t < nb * NH) {
    int ln = t >> 2, h = t & 3;
    int n = node0 + ln;
    if (n < n_nodes) {
      float ern = er[(size_t)n * NH + h];
      float e[DEG];
      float m = -3.4e38f;
#pragma unroll
      for (int j = 0; j < DEG; ++j) {
        int s = src[(size_t)n * DEG + j];
        if (h == 0) {
          s_src[ln * DEG + j] = s;
          __builtin_prefetch(&fs[(size_t)s * colsP], 0, 1); // global_prefetch_b8
        }
        float v = el[(size_t)s * NH + h] + ern;
        v = v > 0.f ? v : NEG_SLOPE * v;
        e[j] = v;
        m = fmaxf(m, v);
      }
      float den = 0.f;
#pragma unroll
      for (int j = 0; j < DEG; ++j) { e[j] = __expf(e[j] - m); den += e[j]; }
      float inv = 1.f / den;
#pragma unroll
      for (int j = 0; j < DEG; ++j) s_alpha[(ln * NH + h) * DEG + j] = e[j] * inv;
    }
  }
  __syncthreads();

  if (t < nb * cols_real) {
    int ln = t / cols_real, c = t - ln * cols_real;
    int n = node0 + ln;
    if (n < n_nodes) {
      int h = c / dout;
      const float* al = &s_alpha[(ln * NH + h) * DEG];
      float acc = 0.f;
#pragma unroll
      for (int j = 0; j < DEG; ++j) {
        int s = s_src[ln * DEG + j];
        acc += al[j] * fs[(size_t)s * colsP + c];
      }
      acc += bias[c];
      if (do_relu) acc = fmaxf(acc, 0.f);
      out[(size_t)n * cols_real + c] = acc;
    }
  }
}

// ----------------- final: mean over heads + log_softmax --------------------
__global__ void gat_final(const float* __restrict__ hout, float* __restrict__ out,
                          int n_nodes) {
  int n = blockIdx.x * blockDim.x + threadIdx.x;
  if (n >= n_nodes) return;
  const float* r = hout + (size_t)n * 188;
  float logit[47];
  float m = -3.4e38f;
#pragma unroll
  for (int c = 0; c < 47; ++c) {
    float s = (r[c] + r[47 + c] + r[94 + c] + r[141 + c]) * 0.25f;
    logit[c] = s;
    m = fmaxf(m, s);
  }
  float den = 0.f;
#pragma unroll
  for (int c = 0; c < 47; ++c) den += __expf(logit[c] - m);
  float lse = __logf(den) + m;
  float* o = out + (size_t)n * 47;
#pragma unroll
  for (int c = 0; c < 47; ++c) o[c] = logit[c] - lse;
}

// ---------------------------------------------------------------------------
extern "C" void kernel_launch(void* const* d_in, const int* in_sizes, int n_in,
                              void* d_out, int out_size, void* d_ws, size_t ws_size,
                              hipStream_t stream) {
  const float* x   = (const float*)d_in[0];
  const int*   src = (const int*)d_in[1];
  // d_in[2] = dst is implicit (repeat(arange(N),16)) -> unused
  const float* Wsrc[3] = {(const float*)d_in[3],  (const float*)d_in[8],  (const float*)d_in[13]};
  const float* Wdst[3] = {(const float*)d_in[4],  (const float*)d_in[9],  (const float*)d_in[14]};
  const float* al[3]   = {(const float*)d_in[5],  (const float*)d_in[10], (const float*)d_in[15]};
  const float* ar[3]   = {(const float*)d_in[6],  (const float*)d_in[11], (const float*)d_in[16]};
  const float* bb[3]   = {(const float*)d_in[7],  (const float*)d_in[12], (const float*)d_in[17]};

  // workspace carve-out (256B aligned)
  char* w = (char*)d_ws;
  auto carve = [&](size_t bytes) { char* p = w; w += (bytes + 255) & ~(size_t)255; return p; };
  _Float16* h16  = (_Float16*)carve((size_t)NN * 128 * sizeof(_Float16)); // 25.6 MB
  float*    fs   = (float*)   carve((size_t)NN * 192 * sizeof(float));    // 76.8 MB
  float*    elb  = (float*)   carve((size_t)NN * NH * sizeof(float));     //  1.6 MB
  float*    erb  = (float*)   carve((size_t)NN * NH * sizeof(float));     //  1.6 MB
  float*    hbuf = (float*)   carve((size_t)NN * 188 * sizeof(float));    // 75.2 MB
  float*    wl   = (float*)   carve(128 * NH * sizeof(float));
  float*    wr   = (float*)   carve(128 * NH * sizeof(float));
  _Float16* Bp   = (_Float16*)carve(32768 * sizeof(_Float16));

  const int fin_[3]    = {128, 64, 64};
  const int dout_[3]   = {16, 16, 47};
  const int cols_[3]   = {64, 64, 188};
  const int colsP_[3]  = {64, 64, 192};
  const int ntiles_[3] = {4, 4, 12};
  const int ksteps_[3] = {4, 2, 2};
  const int mtiles = NN / 16;   // N % 16 == 0

  const float* hin = x;  // fp32 activations of current layer
  for (int l = 0; l < 3; ++l) {
    const int fin = fin_[l], dout = dout_[l], cols = cols_[l];
    const int colsP = colsP_[l], ntiles = ntiles_[l], ksteps = ksteps_[l];

    // activations -> f16 for WMMA A
    int cnt = NN * fin;
    gat_f32_to_f16<<<(cnt + 255) / 256, 256, 0, stream>>>(hin, h16, cnt);
    // pack Wsrc into WMMA-B layout (zero-padded cols)
    int pk = ntiles * ksteps * 512;
    gat_pack_b<<<(pk + 255) / 256, 256, 0, stream>>>(Wsrc[l], Bp, ksteps, ntiles, cols);
    // collapsed attention weights
    int mw = fin * NH;
    gat_make_w<<<(mw + 255) / 256, 256, 0, stream>>>(Wsrc[l], al[l], wl, fin, dout);
    gat_make_w<<<(mw + 255) / 256, 256, 0, stream>>>(Wdst[l], ar[l], wr, fin, dout);
    // fs = h @ Wsrc   (the only big GEMM of the layer)
    int gthreads = mtiles * 32;
    int gblocks  = (gthreads + 255) / 256;
    if (ksteps == 4)
      gat_gemm_wmma<4><<<gblocks, 256, 0, stream>>>(h16, Bp, fs, mtiles, ntiles, colsP);
    else
      gat_gemm_wmma<2><<<gblocks, 256, 0, stream>>>(h16, Bp, fs, mtiles, ntiles, colsP);
    // el / er
    int et = NN * NH;
    gat_edge_logit<<<(et + 255) / 256, 256, 0, stream>>>(hin, wl, elb, NN, fin);
    gat_edge_logit<<<(et + 255) / 256, 256, 0, stream>>>(hin, wr, erb, NN, fin);
    // edge softmax + aggregation (+bias, relu on layers 0/1)
    int nb = (l < 2) ? 4 : 1;
    int ablocks = (NN + nb - 1) / nb;
    gat_attn_agg<<<ablocks, 256, 0, stream>>>(src, elb, erb, fs, bb[l], hbuf,
                                              NN, nb, cols, dout, colsP, l < 2 ? 1 : 0);
    hin = hbuf;
  }

  // mean over heads + log_softmax
  gat_final<<<(NN + 255) / 256, 256, 0, stream>>>(hbuf, (float*)d_out, NN);
}
